// MambaBlockRef0_18124761989162
// MI455X (gfx1250) — matrix operand order repaired and checked by reading, not compiled, not run on hardware
//
#include <hip/hip_runtime.h>

// ---------------------------------------------------------------------------
// Mamba block for MI455X (gfx1250, wave32).
// GEMMs: split-bf16 (hi+lo) V_WMMA_F32_16X16X32_BF16, fp32 accumulate.
// Scan:  fused recompute of dA/dBu (no 537MB tensors), fp32 VALU, lane-per-state
//        with __shfl_xor butterfly reduction (ds permute, wave32).
// ---------------------------------------------------------------------------

#define D_MODEL 1024
#define D_INNER 2048
#define D_STATE 16
#define DT_RANK 64
#define B_SZ    2
#define L_SEQ   2048
#define M_ROWS  (B_SZ * L_SEQ)            /* 4096 */
#define E2      (2 * D_INNER)             /* 4096 */
#define XDBL_C  (DT_RANK + 2 * D_STATE)   /* 96   */

typedef __attribute__((ext_vector_type(16))) __bf16 v16bf;
typedef __attribute__((ext_vector_type(8)))  __bf16 v8bf;
typedef __attribute__((ext_vector_type(8)))  float  v8f;

__device__ __forceinline__ float siluf(float x)     { return x / (1.f + expf(-x)); }
__device__ __forceinline__ float softplusf(float x) { return (x > 20.f) ? x : log1pf(expf(x)); }

// fp32 -> bf16 hi + bf16 lo residual (a ~= hi + lo)
__device__ __forceinline__ void bsplit(float v, __bf16 &hi, __bf16 &lo) {
  hi = (__bf16)v;
  lo = (__bf16)(v - (float)hi);
}

// ---------------------------------------------------------------------------
// LayerNorm over D_MODEL, writing split-bf16 planes for the W_in GEMM.
// One block per row; 256 threads x 4 elements.
// ---------------------------------------------------------------------------
__global__ __launch_bounds__(256) void ln_split_kernel(
    const float* __restrict__ x, const float* __restrict__ g,
    const float* __restrict__ b, __bf16* __restrict__ hhi, __bf16* __restrict__ hlo) {
  const int row = blockIdx.x, tid = threadIdx.x;
  const float* xr = x + (size_t)row * D_MODEL;
  float v[4]; float s = 0.f;
#pragma unroll
  for (int i = 0; i < 4; ++i) { v[i] = xr[tid + i * 256]; s += v[i]; }
  __shared__ float red[256];
  red[tid] = s; __syncthreads();
  for (int o = 128; o > 0; o >>= 1) { if (tid < o) red[tid] += red[tid + o]; __syncthreads(); }
  const float mu = red[0] * (1.f / D_MODEL);
  __syncthreads();
  float s2 = 0.f;
#pragma unroll
  for (int i = 0; i < 4; ++i) { const float d = v[i] - mu; s2 += d * d; }
  red[tid] = s2; __syncthreads();
  for (int o = 128; o > 0; o >>= 1) { if (tid < o) red[tid] += red[tid + o]; __syncthreads(); }
  const float rstd = rsqrtf(red[0] * (1.f / D_MODEL) + 1e-6f);
#pragma unroll
  for (int i = 0; i < 4; ++i) {
    const int c = tid + i * 256;
    const float h = (v[i] - mu) * rstd * g[c] + b[c];
    __bf16 hi, lo; bsplit(h, hi, lo);
    hhi[(size_t)row * D_MODEL + c] = hi;
    hlo[(size_t)row * D_MODEL + c] = lo;
  }
}

// ---------------------------------------------------------------------------
// Dense fp32 -> split-bf16 plane conversion.
// ---------------------------------------------------------------------------
__global__ void split_kernel(const float* __restrict__ in,
                             __bf16* __restrict__ hi, __bf16* __restrict__ lo, int n) {
  const int i = blockIdx.x * 256 + threadIdx.x;
  if (i < n) { __bf16 h, l; bsplit(in[i], h, l); hi[i] = h; lo[i] = l; }
}

// First `cols` columns of a row-major [rows, ld] matrix -> compact split planes.
__global__ void split_cols_kernel(const float* __restrict__ in, int ld, int cols,
                                  __bf16* __restrict__ hi, __bf16* __restrict__ lo, int n) {
  const int i = blockIdx.x * 256 + threadIdx.x;
  if (i < n) {
    const int r = i / cols, c = i - r * cols;
    __bf16 h, l; bsplit(in[(size_t)r * ld + c], h, l);
    hi[i] = h; lo[i] = l;
  }
}

// ---------------------------------------------------------------------------
// Causal depthwise conv (k=4) + bias + SiLU -> u (fp32 for the scan, plus
// split-bf16 planes for the W_x GEMM).
// ---------------------------------------------------------------------------
__global__ void conv_silu_kernel(const float* __restrict__ xz, const float* __restrict__ cw,
                                 const float* __restrict__ cb, float* __restrict__ u,
                                 __bf16* __restrict__ uhi, __bf16* __restrict__ ulo) {
  const int i  = blockIdx.x * 256 + threadIdx.x;   // over M_ROWS * D_INNER
  const int d  = i & (D_INNER - 1);
  const int ml = i >> 11;                          // D_INNER == 2^11
  const int l  = ml & (L_SEQ - 1);
  const int b  = ml >> 11;                         // L_SEQ == 2^11
  float acc = cb[d];
#pragma unroll
  for (int j = 0; j < 4; ++j) {
    const int ls = l - 3 + j;
    if (ls >= 0) acc = fmaf(xz[(size_t)(b * L_SEQ + ls) * E2 + d], cw[d * 4 + j], acc);
  }
  const float uu = siluf(acc);
  u[i] = uu;
  __bf16 h, lo; bsplit(uu, h, lo); uhi[i] = h; ulo[i] = lo;
}

// ---------------------------------------------------------------------------
// Split-bf16 WMMA GEMM:  C[M,N] = A[M,K] * B[N,K]^T  (both K-contiguous).
// One wave per workgroup; wave computes a 16 x (16*NT) tile.
// 3 WMMAs per tile per K-step: Ahi*Bhi + Ahi*Blo + Alo*Bhi (fp32-grade).
// EPI: 0 = plain, 1 = softplus(c + bias[n]), 2 = c + resid[m,n].
// ---------------------------------------------------------------------------
template <int NT, int EPI>
__global__ __launch_bounds__(32) void gemm_wmma_split(
    const __bf16* __restrict__ Ahi, const __bf16* __restrict__ Alo,
    const __bf16* __restrict__ Bhi, const __bf16* __restrict__ Blo,
    int K, const float* __restrict__ bias, const float* __restrict__ resid,
    float* __restrict__ C, int ldc) {
  const int lane = threadIdx.x;
  const int half = lane >> 4, lr = lane & 15;
  const int tm = blockIdx.y * 16;
  const int tn = blockIdx.x * (16 * NT);
  const __bf16* arh = Ahi + (size_t)(tm + lr) * K;
  const __bf16* arl = Alo + (size_t)(tm + lr) * K;
  v8f acc[NT] = {};
  for (int k0 = 0; k0 < K; k0 += 32) {
    // A frag: lane (row=lr, half) holds K = half*16 + 0..15 (contiguous)
    union { v16bf v; v8bf h[2]; } ah, al;
    ah.h[0] = *(const v8bf*)(arh + k0 + half * 16);
    ah.h[1] = *(const v8bf*)(arh + k0 + half * 16 + 8);
    al.h[0] = *(const v8bf*)(arl + k0 + half * 16);
    al.h[1] = *(const v8bf*)(arl + k0 + half * 16 + 8);
    __builtin_prefetch(arh + k0 + 512, 0, 0);   // gfx1250 global_prefetch_b8 (speculative)
#pragma unroll
    for (int nt = 0; nt < NT; ++nt) {
      // B frag: lane (col=lr, half) holds K = k0 + half*16 + 0..15 (contiguous 32B)
      const size_t boff = (size_t)(tn + nt * 16 + lr) * K + k0 + half * 16;
      const v16bf bh = *(const v16bf*)(Bhi + boff);
      const v16bf bl = *(const v16bf*)(Blo + boff);
      acc[nt] = __builtin_amdgcn_wmma_f32_16x16x32_bf16(false, ah.v, false, bh, (short)0, acc[nt], false, false);
      acc[nt] = __builtin_amdgcn_wmma_f32_16x16x32_bf16(false, ah.v, false, bl, (short)0, acc[nt], false, false);
      acc[nt] = __builtin_amdgcn_wmma_f32_16x16x32_bf16(false, al.v, false, bh, (short)0, acc[nt], false, false);
    }
  }
#pragma unroll
  for (int nt = 0; nt < NT; ++nt) {
    const int n = tn + nt * 16 + lr;
#pragma unroll
    for (int r = 0; r < 8; ++r) {
      const int m = tm + half * 8 + r;        // C/D layout: VGPR r -> row half*8 + r
      float v = acc[nt][r];
      if (EPI == 1) v = softplusf(v + bias[n]);
      if (EPI == 2) v += resid[(size_t)m * ldc + n];
      C[(size_t)m * ldc + n] = v;
    }
  }
}

// ---------------------------------------------------------------------------
// Fused selective scan. Lane owns one (d, n) state; dA/dBu recomputed on the
// fly (saves ~1.6 GB of HBM traffic vs materializing them). y reduced over the
// 16 states with a shfl_xor butterfly; epilogue applies +u*Dp and silu(z) gate.
// Block: 256 threads = 16 channels; grid (D_INNER/16, B).
// ---------------------------------------------------------------------------
__global__ __launch_bounds__(256) void scan_kernel(
    const float* __restrict__ dtv, const float* __restrict__ u,
    const float* __restrict__ xdbl, const float* __restrict__ xz,
    const float* __restrict__ A_log, const float* __restrict__ Dp,
    float* __restrict__ yv) {
  const int n    = threadIdx.x & 15;
  const int dloc = threadIdx.x >> 4;
  const int d    = blockIdx.x * 16 + dloc;
  const int b    = blockIdx.y;
  const float An  = -expf(A_log[d * D_STATE + n]);
  const float dpd = Dp[d];
  float s = 0.f;
  for (int t = 0; t < L_SEQ; ++t) {
    const size_t row = (size_t)b * L_SEQ + t;
    const float dt = dtv[row * D_INNER + d];
    const float uu = u  [row * D_INNER + d];
    const float Bn = xdbl[row * XDBL_C + DT_RANK + n];
    const float Cn = xdbl[row * XDBL_C + DT_RANK + D_STATE + n];
    s = fmaf(expf(dt * An), s, dt * uu * Bn);   // h = dA*h + dt*u*B
    float p = s * Cn;
    p += __shfl_xor(p, 1, 16);
    p += __shfl_xor(p, 2, 16);
    p += __shfl_xor(p, 4, 16);
    p += __shfl_xor(p, 8, 16);
    if (n == 0) {
      float y = p + uu * dpd;
      const float zv = xz[row * E2 + D_INNER + d];
      yv[row * D_INNER + d] = y * siluf(zv);
    }
  }
}

// ---------------------------------------------------------------------------
extern "C" void kernel_launch(void* const* d_in, const int* in_sizes, int n_in,
                              void* d_out, int out_size, void* d_ws, size_t ws_size,
                              hipStream_t stream) {
  (void)in_sizes; (void)n_in; (void)out_size;
  const float* x      = (const float*)d_in[0];
  const float* ln_g   = (const float*)d_in[1];
  const float* ln_b   = (const float*)d_in[2];
  const float* W_in   = (const float*)d_in[3];
  const float* conv_w = (const float*)d_in[4];
  const float* conv_b = (const float*)d_in[5];
  const float* W_x    = (const float*)d_in[6];
  const float* W_dt   = (const float*)d_in[7];
  const float* b_dt   = (const float*)d_in[8];
  const float* A_log  = (const float*)d_in[9];
  const float* Dp     = (const float*)d_in[10];
  const float* W_out  = (const float*)d_in[11];
  float* out = (float*)d_out;

  char*  base = (char*)d_ws;
  size_t off  = 0;
  auto carve = [&](size_t bytes) -> char* {
    char* p = base + off;
    off += (bytes + 255) & ~(size_t)255;
    return p;
  };

  __bf16* h_hi   = (__bf16*)carve((size_t)M_ROWS * D_MODEL * 2);
  __bf16* h_lo   = (__bf16*)carve((size_t)M_ROWS * D_MODEL * 2);
  __bf16* Win_hi = (__bf16*)carve((size_t)E2 * D_MODEL * 2);
  __bf16* Win_lo = (__bf16*)carve((size_t)E2 * D_MODEL * 2);
  float*  xz     = (float*) carve((size_t)M_ROWS * E2 * 4);
  float*  u      = (float*) carve((size_t)M_ROWS * D_INNER * 4);
  __bf16* u_hi   = (__bf16*)carve((size_t)M_ROWS * D_INNER * 2);
  __bf16* u_lo   = (__bf16*)carve((size_t)M_ROWS * D_INNER * 2);
  __bf16* Wx_hi  = (__bf16*)carve((size_t)XDBL_C * D_INNER * 2);
  __bf16* Wx_lo  = (__bf16*)carve((size_t)XDBL_C * D_INNER * 2);
  float*  xdbl   = (float*) carve((size_t)M_ROWS * XDBL_C * 4);
  __bf16* dtr_hi = (__bf16*)carve((size_t)M_ROWS * DT_RANK * 2);
  __bf16* dtr_lo = (__bf16*)carve((size_t)M_ROWS * DT_RANK * 2);
  __bf16* Wdt_hi = (__bf16*)carve((size_t)D_INNER * DT_RANK * 2);
  __bf16* Wdt_lo = (__bf16*)carve((size_t)D_INNER * DT_RANK * 2);
  float*  dtv    = (float*) carve((size_t)M_ROWS * D_INNER * 4);
  float*  yv     = (float*) carve((size_t)M_ROWS * D_INNER * 4);
  __bf16* y_hi   = (__bf16*)carve((size_t)M_ROWS * D_INNER * 2);
  __bf16* y_lo   = (__bf16*)carve((size_t)M_ROWS * D_INNER * 2);
  __bf16* Wout_hi= (__bf16*)carve((size_t)D_MODEL * D_INNER * 2);
  __bf16* Wout_lo= (__bf16*)carve((size_t)D_MODEL * D_INNER * 2);
  if (off > ws_size) return;   // workspace too small (~280 MiB needed)

  // 1) LayerNorm -> split-bf16 activations
  ln_split_kernel<<<M_ROWS, 256, 0, stream>>>(x, ln_g, ln_b, h_hi, h_lo);

  // 2) weight splits (deterministic, cheap)
  split_kernel<<<((size_t)E2 * D_MODEL + 255) / 256, 256, 0, stream>>>(W_in, Win_hi, Win_lo, E2 * D_MODEL);
  split_kernel<<<((size_t)XDBL_C * D_INNER + 255) / 256, 256, 0, stream>>>(W_x, Wx_hi, Wx_lo, XDBL_C * D_INNER);
  split_kernel<<<((size_t)D_INNER * DT_RANK + 255) / 256, 256, 0, stream>>>(W_dt, Wdt_hi, Wdt_lo, D_INNER * DT_RANK);
  split_kernel<<<((size_t)D_MODEL * D_INNER + 255) / 256, 256, 0, stream>>>(W_out, Wout_hi, Wout_lo, D_MODEL * D_INNER);

  // 3) xz = h @ W_in^T   (M=4096, N=4096, K=1024)
  gemm_wmma_split<4, 0><<<dim3(E2 / 64, M_ROWS / 16), 32, 0, stream>>>(
      h_hi, h_lo, Win_hi, Win_lo, D_MODEL, nullptr, nullptr, xz, E2);

  // 4) causal depthwise conv + SiLU -> u
  conv_silu_kernel<<<((size_t)M_ROWS * D_INNER) / 256, 256, 0, stream>>>(
      xz, conv_w, conv_b, u, u_hi, u_lo);

  // 5) x_dbl = u @ W_x^T  (M=4096, N=96, K=2048)
  gemm_wmma_split<2, 0><<<dim3(XDBL_C / 32, M_ROWS / 16), 32, 0, stream>>>(
      u_hi, u_lo, Wx_hi, Wx_lo, D_INNER, nullptr, nullptr, xdbl, XDBL_C);

  // 6) dt_r (first 64 cols of x_dbl) -> split planes
  split_cols_kernel<<<((size_t)M_ROWS * DT_RANK + 255) / 256, 256, 0, stream>>>(
      xdbl, XDBL_C, DT_RANK, dtr_hi, dtr_lo, M_ROWS * DT_RANK);

  // 7) dt = softplus(dt_r @ W_dt^T + b_dt)  (M=4096, N=2048, K=64)
  gemm_wmma_split<4, 1><<<dim3(D_INNER / 64, M_ROWS / 16), 32, 0, stream>>>(
      dtr_hi, dtr_lo, Wdt_hi, Wdt_lo, DT_RANK, b_dt, nullptr, dtv, D_INNER);

  // 8) fused selective scan + D-skip + silu(z) gate -> yv
  scan_kernel<<<dim3(D_INNER / 16, B_SZ), 256, 0, stream>>>(
      dtv, u, xdbl, xz, A_log, Dp, yv);

  // 9) yv -> split planes
  split_kernel<<<((size_t)M_ROWS * D_INNER + 255) / 256, 256, 0, stream>>>(
      yv, y_hi, y_lo, M_ROWS * D_INNER);

  // 10) out = x + yv @ W_out^T  (M=4096, N=1024, K=2048), residual in epilogue
  gemm_wmma_split<4, 2><<<dim3(D_MODEL / 64, M_ROWS / 16), 32, 0, stream>>>(
      y_hi, y_lo, Wout_hi, Wout_lo, D_INNER, nullptr, x, out, D_MODEL);
}
